// JointAttention_54494545052293
// MI455X (gfx1250) — matrix operand order repaired and checked by reading, not compiled, not run on hardware
//
#include <hip/hip_runtime.h>
#include <hip/hip_bf16.h>
#include <math.h>

// ---------------------------------------------------------------------------
// Types / helpers for CDNA5 (gfx1250) bf16 WMMA
// ---------------------------------------------------------------------------
typedef __bf16 bhalf;
typedef __bf16 v16bf __attribute__((ext_vector_type(16)));
typedef __bf16 v8bf  __attribute__((ext_vector_type(8)));
typedef float  v8f   __attribute__((ext_vector_type(8)));

__device__ __forceinline__ bhalf f2bh(float f) { return (bhalf)f; }
__device__ __forceinline__ float bh2f(bhalf b) { return (float)b; }

__device__ __forceinline__ v16bf ld_frag(const bhalf* lo, const bhalf* hi) {
  v8bf a = *(const v8bf*)lo;
  v8bf b = *(const v8bf*)hi;
  v16bf r;
#pragma unroll
  for (int i = 0; i < 8; ++i) { r[i] = a[i]; r[i + 8] = b[i]; }
  return r;
}
__device__ __forceinline__ v8f wmma_bf16(v16bf a, v16bf b, v8f c) {
  return __builtin_amdgcn_wmma_f32_16x16x32_bf16(false, a, false, b,
                                                 (short)0, c, false, false);
}
__device__ __forceinline__ v8bf cvt8(float4 a, float4 b) {
  v8bf r;
  r[0] = (bhalf)a.x; r[1] = (bhalf)a.y; r[2] = (bhalf)a.z; r[3] = (bhalf)a.w;
  r[4] = (bhalf)b.x; r[5] = (bhalf)b.y; r[6] = (bhalf)b.z; r[7] = (bhalf)b.w;
  return r;
}
// Generic->LDS offset: low 32 bits of a generic LDS address are the LDS offset.
__device__ __forceinline__ unsigned lds_off(const void* p) {
  return (unsigned)(uintptr_t)p;
}
// Async DMA: copy 32B / 64B per lane, global -> LDS (ASYNCcnt tracked).
// INST_OFFSET is added to both the LDS and global addresses (ISA 8.4.4).
__device__ __forceinline__ void async_copy32(unsigned lds, unsigned voff,
                                             const void* saddr) {
  asm volatile(
      "global_load_async_to_lds_b128 %0, %1, %2 offset:0\n\t"
      "global_load_async_to_lds_b128 %0, %1, %2 offset:16"
      :: "v"(lds), "v"(voff), "s"(saddr) : "memory");
}
__device__ __forceinline__ void async_copy64(unsigned lds, unsigned voff,
                                             const void* saddr) {
  asm volatile(
      "global_load_async_to_lds_b128 %0, %1, %2 offset:0\n\t"
      "global_load_async_to_lds_b128 %0, %1, %2 offset:16\n\t"
      "global_load_async_to_lds_b128 %0, %1, %2 offset:32\n\t"
      "global_load_async_to_lds_b128 %0, %1, %2 offset:48"
      :: "v"(lds), "v"(voff), "s"(saddr) : "memory");
}
__device__ __forceinline__ void wait_async0() {
  asm volatile("s_wait_asynccnt 0x0" ::: "memory");
}

// ---------------------------------------------------------------------------
// Elementwise f32 -> bf16 (8 elements/thread)
// ---------------------------------------------------------------------------
__global__ __launch_bounds__(256) void cvt_bf16_kernel(
    const float* __restrict__ in, bhalf* __restrict__ out) {
  const size_t i = ((size_t)blockIdx.x * 256 + threadIdx.x) * 8;
  float4 a = *(const float4*)(in + i);
  float4 b = *(const float4*)(in + i + 4);
  *(v8bf*)(out + i) = cvt8(a, b);
}

// ---------------------------------------------------------------------------
// Tiled transpose + convert: in f32 [R][C] -> out bf16 [C][R]
// ---------------------------------------------------------------------------
__global__ __launch_bounds__(256) void transpose_cvt_kernel(
    const float* __restrict__ in, bhalf* __restrict__ out, int R, int C) {
  __shared__ float tile[32][33];
  const int bx = blockIdx.x * 32;  // column tile
  const int by = blockIdx.y * 32;  // row tile
  const int tx = threadIdx.x & 31;
  const int ty = threadIdx.x >> 5;  // 8 rows per pass
#pragma unroll
  for (int j = 0; j < 32; j += 8)
    tile[ty + j][tx] = in[(size_t)(by + ty + j) * C + bx + tx];
  __syncthreads();
#pragma unroll
  for (int j = 0; j < 32; j += 8)
    out[(size_t)(bx + ty + j) * R + by + tx] = f2bh(tile[tx][ty + j]);
}

// ---------------------------------------------------------------------------
// bf16 WMMA GEMM with async-DMA double-buffered staging.
//   C[MxN] = A[MxK](bf16,row) * Bt[NxK](bf16, B stored transposed)
// Workgroup: 128 threads (4 waves). Tile: 64(M) x 64(N), K step 32.
// ---------------------------------------------------------------------------
template <bool OUT_F32>
__global__ __launch_bounds__(128) void gemm_bf16_kernel(
    const bhalf* __restrict__ A, int lda, const bhalf* __restrict__ Bt,
    int ldb, void* __restrict__ Cp, int ldc, int K) {
  __shared__ bhalf As[2][64 * 32];  // [row][k], row stride 64B
  __shared__ bhalf Bs[2][64 * 32];  // [n][k],   row stride 64B

  const int tid  = threadIdx.x;
  const int lane = tid & 31;
  const int wave = tid >> 5;
  const int n0 = blockIdx.x * 64;
  const int m0 = blockIdx.y * 64;

  const int l16  = lane & 15;
  const int half = lane >> 4;
  const int off8 = half * 8;
  const int klo  = half * 16;

  // staging: 64 rows, 2 threads/row, 16 bf16 (32B) each
  const int srow = tid >> 1;
  const int sseg = (tid & 1) * 16;
  const unsigned ldsA[2] = {lds_off(&As[0][srow * 32 + sseg]),
                            lds_off(&As[1][srow * 32 + sseg])};
  const unsigned ldsB[2] = {lds_off(&Bs[0][srow * 32 + sseg]),
                            lds_off(&Bs[1][srow * 32 + sseg])};
  const unsigned aRow = (unsigned)((m0 + srow) * lda + sseg) * 2u;
  const unsigned bRow = (unsigned)((n0 + srow) * ldb + sseg) * 2u;

  v8f acc[4];
#pragma unroll
  for (int nt = 0; nt < 4; ++nt)
#pragma unroll
    for (int i = 0; i < 8; ++i) acc[nt][i] = 0.0f;

  // prologue: DMA tile 0 into buffer 0
  async_copy32(ldsA[0], aRow, A);
  async_copy32(ldsB[0], bRow, Bt);

  const int T = K >> 5;
#pragma unroll 1
  for (int t = 0; t < T; ++t) {
    wait_async0();
    __syncthreads();  // tile t resident in buf (everyone)
    const int buf = t & 1;
    if (t + 1 < T) {  // DMA tile t+1 into the other buffer
      const unsigned koff = (unsigned)((t + 1) << 5) * 2u;
      async_copy32(ldsA[buf ^ 1], aRow + koff, A);
      async_copy32(ldsB[buf ^ 1], bRow + koff, Bt);
    }
    const int arow = wave * 16 + l16;
    v16bf a = ld_frag(&As[buf][arow * 32 + off8],
                      &As[buf][arow * 32 + 16 + off8]);
#pragma unroll
    for (int nt = 0; nt < 4; ++nt) {
      const int bn = nt * 16 + l16;
      v16bf b = ld_frag(&Bs[buf][bn * 32 + off8], &Bs[buf][bn * 32 + 16 + off8]);
      acc[nt] = wmma_bf16(a, b, acc[nt]);
    }
  }

#pragma unroll
  for (int nt = 0; nt < 4; ++nt) {
#pragma unroll
    for (int r = 0; r < 8; ++r) {
      const int row = m0 + wave * 16 + r + half * 8;
      const int col = n0 + nt * 16 + l16;
      if (OUT_F32)
        ((float*)Cp)[(size_t)row * ldc + col] = acc[nt][r];
      else
        ((bhalf*)Cp)[(size_t)row * ldc + col] = f2bh(acc[nt][r]);
    }
  }
}

// ---------------------------------------------------------------------------
// QKV prep: RMS-norm(q,k) + RoPE (positions restart per modality).
// Q gets the 1/sqrt(d) scale folded in. Q,K: [h][S][64]; Vt: [h][64][S].
// ---------------------------------------------------------------------------
__global__ __launch_bounds__(256) void qkv_prep_kernel(
    const bhalf* __restrict__ qkv, const float* __restrict__ gq0,
    const float* __restrict__ gk0, const float* __restrict__ gq1,
    const float* __restrict__ gk1, bhalf* __restrict__ Q,
    bhalf* __restrict__ Kx, bhalf* __restrict__ Vt) {
  const int S = 3072, H = 24;
  const int idx = blockIdx.x * 256 + threadIdx.x;
  const int s = idx / H;
  const int h = idx % H;
  const int mod = (s >= 512);
  const float pos = (float)(mod ? s - 512 : s);
  const float* gq = mod ? gq1 : gq0;
  const float* gk = mod ? gk1 : gk0;

  const bhalf* qrow = qkv + (size_t)s * 4608 + h * 64;
  const bhalf* krow = qrow + 1536;
  const bhalf* vrow = qrow + 3072;

  float ssq_q = 0.f, ssq_k = 0.f;
#pragma unroll 1
  for (int d = 0; d < 64; ++d) {
    float q = bh2f(qrow[d]), k = bh2f(krow[d]);
    ssq_q += q * q;
    ssq_k += k * k;
  }
  const float rq = rsqrtf(ssq_q * (1.0f / 64.0f) + 1e-6f);
  const float rk = rsqrtf(ssq_k * (1.0f / 64.0f) + 1e-6f);

  const float kfreq = logf(10000.0f) * (1.0f / 32.0f);
#pragma unroll 1
  for (int d = 0; d < 64; ++d) {
    const int i = d & 31;
    const float ang = pos * __expf(-(float)i * kfreq);
    const float c = __cosf(ang), sn = __sinf(ang);
    const int dp = (d < 32) ? d + 32 : d - 32;
    const float sgn = (d < 32) ? -1.0f : 1.0f;

    const float qd = bh2f(qrow[d]) * rq * gq[d];
    const float qp = bh2f(qrow[dp]) * rq * gq[dp];
    const float kd = bh2f(krow[d]) * rk * gk[d];
    const float kp = bh2f(krow[dp]) * rk * gk[dp];

    const size_t o = ((size_t)h * S + s) * 64 + d;
    Q[o]  = f2bh((qd * c + sgn * qp * sn) * 0.125f);
    Kx[o] = f2bh(kd * c + sgn * kp * sn);
    Vt[((size_t)h * 64 + d) * S + s] = vrow[d];
  }
}

// ---------------------------------------------------------------------------
// Flash attention, no-max softmax (logits bounded ~8 after RMS-norm+scale),
// async-DMA double-buffered K/V staging. Grid(S/64, heads), 128 thr (4 waves).
// Output written bf16 (feeds the bf16 out-projection GEMM).
// ---------------------------------------------------------------------------
__global__ __launch_bounds__(128) void attn_kernel(
    const bhalf* __restrict__ Q, const bhalf* __restrict__ Km,
    const bhalf* __restrict__ Vt, bhalf* __restrict__ Out) {
  __shared__ bhalf Ks[2][64 * 64];    // [kpos][dim], row stride 128B
  __shared__ bhalf Vs[2][64 * 64];    // [dim][kpos], row stride 128B
  __shared__ bhalf Ps[4][16 * 64];    // per-wave P scratch

  const int S = 3072;
  const int tid = threadIdx.x, lane = tid & 31, wave = tid >> 5;
  const int h = blockIdx.y;
  const int qbase = blockIdx.x * 64 + wave * 16;
  const int l16 = lane & 15, half = lane >> 4;
  const int off8 = half * 8, klo = half * 16;

  const bhalf* Qh  = Q + (size_t)h * S * 64;
  const bhalf* Kh  = Km + (size_t)h * S * 64;
  const bhalf* Vth = Vt + (size_t)h * 64 * S;  // [dim][S]

  const bhalf* qp = Qh + (size_t)(qbase + l16) * 64;
  const v16bf aq0 = ld_frag(qp + off8, qp + 16 + off8);
  const v16bf aq1 = ld_frag(qp + 32 + off8, qp + 48 + off8);

  v8f o[4];
  float lsum[8];
#pragma unroll
  for (int dt = 0; dt < 4; ++dt)
#pragma unroll
    for (int i = 0; i < 8; ++i) o[dt][i] = 0.0f;
#pragma unroll
  for (int r = 0; r < 8; ++r) lsum[r] = 0.0f;

  // staging: 64 rows, 2 threads/row, 32 bf16 (64B) each
  const int srow = tid >> 1;
  const int sseg = (tid & 1) * 32;
  const unsigned ldsK[2] = {lds_off(&Ks[0][srow * 64 + sseg]),
                            lds_off(&Ks[1][srow * 64 + sseg])};
  const unsigned ldsV[2] = {lds_off(&Vs[0][srow * 64 + sseg]),
                            lds_off(&Vs[1][srow * 64 + sseg])};
  const unsigned kRow = (unsigned)(srow * 64 + sseg) * 2u;   // + kb*128B
  const unsigned vRow = (unsigned)(srow * S + sseg) * 2u;    // + kb*2B

  // prologue: DMA key-block 0 into buffer 0
  async_copy64(ldsK[0], kRow, Kh);
  async_copy64(ldsV[0], vRow, Vth);

#pragma unroll 1
  for (int t = 0; t < S / 64; ++t) {
    wait_async0();
    __syncthreads();
    const int buf = t & 1;
    if (t + 1 < S / 64) {
      async_copy64(ldsK[buf ^ 1], kRow + (unsigned)(t + 1) * 64 * 128u, Kh);
      async_copy64(ldsV[buf ^ 1], vRow + (unsigned)(t + 1) * 64 * 2u, Vth);
    }

    // ---- scores: S[16x64] = Q(16x64) . K^T ----
    v8f s[4];
#pragma unroll
    for (int nt = 0; nt < 4; ++nt) {
#pragma unroll
      for (int i = 0; i < 8; ++i) s[nt][i] = 0.0f;
      const bhalf* kr = &Ks[buf][(size_t)(nt * 16 + l16) * 64];
      s[nt] = wmma_bf16(aq0, ld_frag(kr + off8, kr + 16 + off8), s[nt]);
      s[nt] = wmma_bf16(aq1, ld_frag(kr + 32 + off8, kr + 48 + off8), s[nt]);
    }

    // ---- exp, per-lane partial denominator, P -> LDS ----
    {
      bhalf* pw = &Ps[wave][0];
#pragma unroll
      for (int nt = 0; nt < 4; ++nt)
#pragma unroll
        for (int r = 0; r < 8; ++r) {
          const float e = __expf(s[nt][r]);
          lsum[r] += e;
          pw[(r + half * 8) * 64 + nt * 16 + l16] = f2bh(e);
        }
    }
    asm volatile("s_wait_dscnt 0x0" ::: "memory");  // same-wave LDS RAW

    const bhalf* pr = &Ps[wave][(size_t)l16 * 64];
    const v16bf ap0 = ld_frag(pr + off8, pr + 16 + off8);
    const v16bf ap1 = ld_frag(pr + 32 + off8, pr + 48 + off8);

    // ---- O += P(16x64) . V(64x64) ----
#pragma unroll
    for (int dt = 0; dt < 4; ++dt) {
      const bhalf* vp = &Vs[buf][(size_t)(dt * 16 + l16) * 64];
      o[dt] = wmma_bf16(ap0, ld_frag(vp + off8, vp + 16 + off8), o[dt]);
      o[dt] = wmma_bf16(ap1, ld_frag(vp + 32 + off8, vp + 48 + off8), o[dt]);
    }
  }

  // ---- one cross-lane reduction of the denominators (16-lane halves) ----
#pragma unroll
  for (int r = 0; r < 8; ++r) {
    lsum[r] += __shfl_xor(lsum[r], 1, 32);
    lsum[r] += __shfl_xor(lsum[r], 2, 32);
    lsum[r] += __shfl_xor(lsum[r], 4, 32);
    lsum[r] += __shfl_xor(lsum[r], 8, 32);
  }

#pragma unroll
  for (int r = 0; r < 8; ++r) {
    const float inv = 1.0f / lsum[r];
    const int row = qbase + r + half * 8;
#pragma unroll
    for (int dt = 0; dt < 4; ++dt) {
      const int col = h * 64 + dt * 16 + l16;
      Out[(size_t)row * 1536 + col] = f2bh(o[dt][r] * inv);
    }
  }
}

// ---------------------------------------------------------------------------
// Launch
// ---------------------------------------------------------------------------
extern "C" void kernel_launch(void* const* d_in, const int* in_sizes, int n_in,
                              void* d_out, int out_size, void* d_ws,
                              size_t ws_size, hipStream_t stream) {
  const float* x0    = (const float*)d_in[0];
  const float* x1    = (const float*)d_in[1];
  const float* wqkv0 = (const float*)d_in[2];
  const float* wqkv1 = (const float*)d_in[3];
  const float* gq0   = (const float*)d_in[4];
  const float* gk0   = (const float*)d_in[5];
  const float* gq1   = (const float*)d_in[6];
  const float* gk1   = (const float*)d_in[7];
  const float* wout0 = (const float*)d_in[8];
  const float* wout1 = (const float*)d_in[9];

  char* ws = (char*)d_ws;
  bhalf* xb      = (bhalf*)(ws);                   // [3072][1536]
  bhalf* wqkv0T  = (bhalf*)(ws + 9437184);         // [4608][1536]
  bhalf* wqkv1T  = (bhalf*)(ws + 23592960);        // [4608][1536]
  bhalf* wout0T  = (bhalf*)(ws + 37748736);        // [1536][1536]
  bhalf* wout1T  = (bhalf*)(ws + 42467328);        // [1536][1536]
  bhalf* qkv     = (bhalf*)(ws + 47185920);        // [3072][4608]
  bhalf* Q       = (bhalf*)(ws + 75497472);        // [24][3072][64]
  bhalf* Kx      = (bhalf*)(ws + 84934656);        // [24][3072][64]
  bhalf* Vt      = (bhalf*)(ws + 94371840);        // [24][64][3072]
  bhalf* attn    = (bhalf*)(ws + 103809024);       // [3072][1536]

  // ---- precision prep: bf16 activations, transposed bf16 weights ----
  cvt_bf16_kernel<<<384, 256, 0, stream>>>(x0, xb);
  cvt_bf16_kernel<<<1920, 256, 0, stream>>>(x1, xb + (size_t)512 * 1536);
  transpose_cvt_kernel<<<dim3(144, 48), 256, 0, stream>>>(wqkv0, wqkv0T, 1536, 4608);
  transpose_cvt_kernel<<<dim3(144, 48), 256, 0, stream>>>(wqkv1, wqkv1T, 1536, 4608);
  transpose_cvt_kernel<<<dim3(48, 48), 256, 0, stream>>>(wout0, wout0T, 1536, 1536);
  transpose_cvt_kernel<<<dim3(48, 48), 256, 0, stream>>>(wout1, wout1T, 1536, 1536);

  const dim3 blk(128);

  // ---- QKV projections ----
  gemm_bf16_kernel<false><<<dim3(72, 8), blk, 0, stream>>>(
      xb, 1536, wqkv0T, 1536, qkv, 4608, 1536);
  gemm_bf16_kernel<false><<<dim3(72, 40), blk, 0, stream>>>(
      xb + (size_t)512 * 1536, 1536, wqkv1T, 1536,
      qkv + (size_t)512 * 4608, 4608, 1536);

  // ---- RMS-norm + RoPE + repack ----
  qkv_prep_kernel<<<288, 256, 0, stream>>>(qkv, gq0, gk0, gq1, gk1, Q, Kx, Vt);

  // ---- joint attention ----
  attn_kernel<<<dim3(48, 24), blk, 0, stream>>>(Q, Kx, Vt, attn);

  // ---- output projections (f32 into d_out) ----
  gemm_bf16_kernel<true><<<dim3(24, 8), blk, 0, stream>>>(
      attn, 1536, wout0T, 1536, (float*)d_out, 1536, 1536);
  gemm_bf16_kernel<true><<<dim3(24, 40), blk, 0, stream>>>(
      attn + (size_t)512 * 1536, 1536, wout1T, 1536,
      (float*)d_out + (size_t)512 * 1536, 1536, 1536);
}
